// UpXiPredictorAttnPos_Query_9182640078993
// MI455X (gfx1250) — matrix operand order repaired and checked by reading, not compiled, not run on hardware
//
#include <hip/hip_runtime.h>
#include <hip/hip_bf16.h>
#include <math.h>

#define Bb 4
#define Nn 8192
#define KP 1024
#define Cc 256
#define NFq 9
#define Hh 256

__constant__ float kPI = 3.14159265358979323846f;

constexpr float TAUc = 1.0f;
constexpr float PRIOR_STR = 2.0f;
constexpr float PRIOR_FLOOR = 1e-6f;
constexpr float XI_SCALE = 1.5f;
constexpr float POS_WEIGHT = 1.0f;
constexpr float ANG_SCALE = 10.0f;
constexpr float EPSc = 1e-8f;

typedef __attribute__((ext_vector_type(16))) __bf16 v16bf;
typedef __attribute__((ext_vector_type(8)))  float  v8f;

__device__ inline unsigned short f2bf(float f) {
    unsigned int u = __float_as_uint(f);
    unsigned int r = (u + 0x7FFFu + ((u >> 16) & 1u)) >> 16;
    return (unsigned short)r;
}
__device__ inline float bf2f(unsigned short h) {
    return __uint_as_float(((unsigned int)h) << 16);
}

// A-matrix fragment (16x32 bf16): lane L holds row M=L%16; K offsets
// {b0..b0+7} and {b0+16..b0+23}, b0 = (L>=16)?8:0   (ISA 7.12.2)
__device__ inline v16bf load_a_frag(const unsigned short* rowbase, int k0) {
    int lane = threadIdx.x & 31;
    int b0 = k0 + ((lane & 16) ? 8 : 0);
    union { uint4 u[2]; v16bf v; } U;
    U.u[0] = *(const uint4*)(rowbase + b0);
    U.u[1] = *(const uint4*)(rowbase + b0 + 16);
    return U.v;
}
// B-matrix fragment (32x16 bf16): lane L holds column N=L%16; K offsets
// {b0..b0+7} and {b0+16..b0+23}, b0 = (L>=16)?8:0. rowbase = row of W (N-index).
__device__ inline v16bf load_b_frag(const unsigned short* rowbase, int k0) {
    int lane = threadIdx.x & 31;
    int b0 = k0 + ((lane & 16) ? 8 : 0);
    union { uint4 u[2]; v16bf v; } U;
    U.u[0] = *(const uint4*)(rowbase + b0);
    U.u[1] = *(const uint4*)(rowbase + b0 + 16);
    return U.v;
}

// Async global->LDS 16-byte copy (GLOBAL_LOAD_ASYNC_TO_LDS_B128, ASYNCcnt).
// VDST = VGPR holding LDS byte address (low 32 bits of generic shared ptr),
// VADDR = 64-bit global address, GV mode (no SADDR).
__device__ inline void async_copy_b128(unsigned short* lds_dst, const unsigned short* gsrc) {
    unsigned int lds_addr = (unsigned int)(unsigned long long)lds_dst;
    asm volatile("global_load_async_to_lds_b128 %0, %1, off"
                 :: "v"(lds_addr), "v"(gsrc) : "memory");
}
__device__ inline void wait_asynccnt0() {
    asm volatile("s_wait_asynccnt 0x0" ::: "memory");
}

// ---------------- elementwise / LN kernels ----------------

__global__ void cast_bf16_kernel(const float* __restrict__ X, unsigned short* __restrict__ Y, int n) {
    int i = blockIdx.x * blockDim.x + threadIdx.x;
    if (i < n) Y[i] = f2bf(X[i]);
}

__global__ void zero_kernel(float* p, int n) {
    int i = blockIdx.x * blockDim.x + threadIdx.x;
    if (i < n) p[i] = 0.0f;
}

// LayerNorm over C=256 then cast to bf16. One row per 256-thread block.
__global__ __launch_bounds__(256) void ln_cast_kernel(const float* __restrict__ X,
                                                      const float* __restrict__ g,
                                                      const float* __restrict__ bta,
                                                      unsigned short* __restrict__ Y) {
    __shared__ float red[256];
    int row = blockIdx.x, t = threadIdx.x;
    float x = X[(size_t)row * Cc + t];
    red[t] = x; __syncthreads();
    #pragma unroll
    for (int s = 128; s > 0; s >>= 1) { if (t < s) red[t] += red[t + s]; __syncthreads(); }
    float mean = red[0] * (1.0f / Cc);
    __syncthreads();
    float d = x - mean;
    red[t] = d * d; __syncthreads();
    #pragma unroll
    for (int s = 128; s > 0; s >>= 1) { if (t < s) red[t] += red[t + s]; __syncthreads(); }
    float var = red[0] * (1.0f / Cc);
    float y = d * rsqrtf(var + 1e-5f) * g[t] + bta[t];
    Y[(size_t)row * Cc + t] = f2bf(y);
}

// ---------------- WMMA GEMM: D[M,256] = act(A_bf16 @ W_bf16^T + bias) ----------------
template <int ACT, int OUT_BF16>
__global__ __launch_bounds__(256) void wmma_gemm256(const unsigned short* __restrict__ A,
                                                    const unsigned short* __restrict__ W,
                                                    const float* __restrict__ bias,
                                                    float* __restrict__ outF,
                                                    unsigned short* __restrict__ outB,
                                                    int Mrows) {
    int wave = threadIdx.x >> 5;
    int lane = threadIdx.x & 31;
    int mtile = blockIdx.x * 8 + wave;
    if (mtile * 16 >= Mrows) return;
    int mrow = mtile * 16 + (lane & 15);
    const unsigned short* arow = A + (size_t)mrow * Cc;
    v16bf afr[8];
    #pragma unroll
    for (int kt = 0; kt < 8; ++kt) afr[kt] = load_a_frag(arow, kt * 32);
    for (int jt = 0; jt < 16; ++jt) {
        const unsigned short* wrow = W + (size_t)(jt * 16 + (lane & 15)) * Cc;
        v16bf bfr[8];
        #pragma unroll
        for (int kt = 0; kt < 8; ++kt) bfr[kt] = load_b_frag(wrow, kt * 32);
        v8f acc = {};
        #pragma unroll
        for (int kt = 0; kt < 8; ++kt)
            acc = __builtin_amdgcn_wmma_f32_16x16x32_bf16(false, afr[kt], false, bfr[kt],
                                                          (short)0, acc, false, false);
        int col = jt * 16 + (lane & 15);
        float bb = bias ? bias[col] : 0.0f;
        #pragma unroll
        for (int r = 0; r < 8; ++r) {
            int m = (lane < 16) ? r : (r + 8);
            size_t row = (size_t)mtile * 16 + m;
            float v = acc[r] + bb;
            if (ACT == 1) v = v / (1.0f + __expf(-v));
            if (OUT_BF16) outB[row * Cc + col] = f2bf(v);
            else          outF[row * Cc + col] = v;
        }
    }
}

// k = LN(s_parent)@Wk.T (already in kf) + mu_k@Wmu.T, cast to bf16
__global__ __launch_bounds__(256) void k_finish_kernel(const float* __restrict__ kf,
                                                       const float* __restrict__ mu_k,
                                                       const float* __restrict__ Wmu,
                                                       unsigned short* __restrict__ kbf) {
    int row = blockIdx.x, c = threadIdx.x;
    float v = kf[(size_t)row * Cc + c];
    const float* mu = mu_k + (size_t)row * 3;
    v += mu[0] * Wmu[c * 3 + 0] + mu[1] * Wmu[c * 3 + 1] + mu[2] * Wmu[c * 3 + 2];
    kbf[(size_t)row * Cc + c] = f2bf(v);
}

// ---------------- fused WMMA attention logits + mask + prior ----------------
// grid: (N/16/8, B), block 256 (8 waves, one 16-row q tile each).
// k-tiles double-buffered in LDS via GLOBAL_LOAD_ASYNC_TO_LDS_B128.
__global__ __launch_bounds__(256) void attn_logits_kernel(const unsigned short* __restrict__ qbf,
                                                          const unsigned short* __restrict__ kbf,
                                                          const float* __restrict__ A_soft,
                                                          const float* __restrict__ node_mask,
                                                          const float* __restrict__ mask_parent,
                                                          float* __restrict__ logits) {
    __shared__ __align__(16) unsigned short kt_lds[2][16 * Cc];   // 2 x 8 KB
    int b = blockIdx.y;
    int wave = threadIdx.x >> 5;
    int lane = threadIdx.x & 31;
    int mtile = blockIdx.x * 8 + wave;
    int mrow = mtile * 16 + (lane & 15);
    const unsigned short* qrow = qbf + (size_t)(b * Nn + mrow) * Cc;
    v16bf afr[8];
    #pragma unroll
    for (int kt = 0; kt < 8; ++kt) afr[kt] = load_a_frag(qrow, kt * 32);

    // loop-invariant per-row node mask (8 rows per lane)
    float mNr[8];
    #pragma unroll
    for (int r = 0; r < 8; ++r) {
        int m = (lane < 16) ? r : (r + 8);
        mNr[r] = node_mask[b * Nn + mtile * 16 + m];
    }

    // stage tile 0 asynchronously: 512 x 16B, 2 per thread
    {
        const unsigned short* src = kbf + (size_t)(b * KP) * Cc;
        async_copy_b128(&kt_lds[0][threadIdx.x * 8], src + threadIdx.x * 8);
        async_copy_b128(&kt_lds[0][(threadIdx.x + 256) * 8], src + (threadIdx.x + 256) * 8);
    }

    const float invscale = 1.0f / (16.0f * TAUc);  // sqrt(256)=16
    int cur = 0;
    for (int kt0 = 0; kt0 < KP / 16; ++kt0) {
        wait_asynccnt0();      // own async stores to LDS complete
        __syncthreads();       // => every wave's stores complete
        if (kt0 + 1 < KP / 16) {   // prefetch next tile into other buffer
            const unsigned short* src = kbf + (size_t)(b * KP + (kt0 + 1) * 16) * Cc;
            async_copy_b128(&kt_lds[cur ^ 1][threadIdx.x * 8], src + threadIdx.x * 8);
            async_copy_b128(&kt_lds[cur ^ 1][(threadIdx.x + 256) * 8], src + (threadIdx.x + 256) * 8);
        }

        const unsigned short* wrow = &kt_lds[cur][(lane & 15) * Cc];
        v16bf bfr[8];
        #pragma unroll
        for (int kt = 0; kt < 8; ++kt) bfr[kt] = load_b_frag(wrow, kt * 32);
        v8f acc = {};
        #pragma unroll
        for (int kt = 0; kt < 8; ++kt)
            acc = __builtin_amdgcn_wmma_f32_16x16x32_bf16(false, afr[kt], false, bfr[kt],
                                                          (short)0, acc, false, false);

        int kk = kt0 * 16 + (lane & 15);
        float mK = mask_parent[b * KP + kk];
        #pragma unroll
        for (int r = 0; r < 8; ++r) {
            int m = (lane < 16) ? r : (r + 8);
            int n = mtile * 16 + m;
            float mN = mNr[r];
            float lg = acc[r] * invscale;
            if (mK < 0.5f) lg = -10000.0f;
            if (mN < 0.5f) lg = -10000.0f;
            float prior = fmaxf(A_soft[(size_t)(b * Nn + n) * KP + kk] * mN * mK, PRIOR_FLOOR);
            lg += PRIOR_STR * __logf(prior);
            logits[(size_t)(b * Nn + n) * KP + kk] = lg;
        }
        cur ^= 1;
    }
}

// ---------------- softmax over K=1024 (one wave per row, in place) + argmax ----------------
__global__ __launch_bounds__(256) void softmax_kernel(float* __restrict__ logits,
                                                      const float* __restrict__ node_mask,
                                                      int* __restrict__ pidx) {
    int wave = threadIdx.x >> 5;
    int lane = threadIdx.x & 31;
    size_t row = (size_t)blockIdx.x * 8 + wave;
    float* L = logits + row * KP;
    float vals[32];
    float mx = -3.0e38f;
    #pragma unroll
    for (int i = 0; i < 32; ++i) { vals[i] = L[lane + i * 32]; mx = fmaxf(mx, vals[i]); }
    #pragma unroll
    for (int off = 16; off > 0; off >>= 1) mx = fmaxf(mx, __shfl_xor(mx, off, 32));
    float sum = 0.0f;
    float bestv = -3.0e38f; int bestc = 0;
    #pragma unroll
    for (int i = 0; i < 32; ++i) {
        int col = lane + i * 32;
        if (vals[i] > bestv) { bestv = vals[i]; bestc = col; }
        float e = __expf(vals[i] - mx);
        vals[i] = e; sum += e;
    }
    #pragma unroll
    for (int off = 16; off > 0; off >>= 1) sum += __shfl_xor(sum, off, 32);
    #pragma unroll
    for (int off = 16; off > 0; off >>= 1) {
        float ov = __shfl_xor(bestv, off, 32);
        int   oc = __shfl_xor(bestc, off, 32);
        if (ov > bestv || (ov == bestv && oc < bestc)) { bestv = ov; bestc = oc; }
    }
    float mN = node_mask[row];
    float inv = mN / fmaxf(sum, 1e-30f);
    #pragma unroll
    for (int i = 0; i < 32; ++i) L[lane + i * 32] = vals[i] * inv;
    if (lane == 0) pidx[row] = bestc;
}

// total[b,k] = sum_n w[b,n,k]
__global__ __launch_bounds__(256) void col_total_kernel(const float* __restrict__ w,
                                                        float* __restrict__ total) {
    int b = blockIdx.y;
    int k = blockIdx.x * 256 + threadIdx.x;
    float s = 0.0f;
    for (int n = 0; n < Nn; ++n) s += w[(size_t)(b * Nn + n) * KP + k];
    total[b * KP + k] = s;
}

// pos01[b,n] = sum_k w*cum/total  (running scan over n; block handles 256 k-columns)
__global__ __launch_bounds__(256) void pos_scan_kernel(const float* __restrict__ w,
                                                       const float* __restrict__ total,
                                                       float* __restrict__ pos01) {
    __shared__ float red[256];
    int b = blockIdx.y;
    int k = blockIdx.x * 256 + threadIdx.x;
    float tot = fmaxf(total[b * KP + k], EPSc);
    float inv = 1.0f / tot;
    float cum = 0.0f;
    for (int n = 0; n < Nn; ++n) {
        float wv = w[(size_t)(b * Nn + n) * KP + k];
        cum += wv;
        red[threadIdx.x] = wv * cum * inv;
        __syncthreads();
        #pragma unroll
        for (int s = 128; s > 0; s >>= 1) { if (threadIdx.x < s) red[threadIdx.x] += red[threadIdx.x + s]; __syncthreads(); }
        if (threadIdx.x == 0) atomicAdd(&pos01[b * Nn + n], red[0]);
        __syncthreads();
    }
}

// s_pi = gather(s_parent, parent_idx)*mN + POS_WEIGHT*pos_emb, all * mN
__global__ __launch_bounds__(256) void spi_kernel(const float* __restrict__ pos01,
                                                  const float* __restrict__ posW,
                                                  const float* __restrict__ posb,
                                                  const float* __restrict__ s_parent,
                                                  const int* __restrict__ pidx,
                                                  const float* __restrict__ node_mask,
                                                  float* __restrict__ s_pi) {
    __shared__ float feats[2 * NFq];
    int nn = blockIdx.x;           // flat b*N+n
    int c = threadIdx.x;
    float mN = node_mask[nn];
    if (threadIdx.x < NFq) {
        float p = pos01[nn] * mN;
        float ang = p * exp2f((float)threadIdx.x) * kPI;
        feats[threadIdx.x]       = __sinf(ang);
        feats[threadIdx.x + NFq] = __cosf(ang);
    }
    __syncthreads();
    float pe = posb[c];
    #pragma unroll
    for (int j = 0; j < 2 * NFq; ++j) pe += feats[j] * posW[c * 2 * NFq + j];
    pe *= mN;
    int b = nn / Nn;
    int idx = pidx[nn];
    float sp = s_parent[(size_t)(b * KP + idx) * Cc + c];
    s_pi[(size_t)nn * Cc + c] = (sp * mN + POS_WEIGHT * pe) * mN;
}

// final head: xi = tanh((z@mW3.T+mb3)*mN)*XI_SCALE; x_hat=(mu_i + R_i@(xi*s_i))*mN*ANG
__global__ __launch_bounds__(128) void final_kernel(const unsigned short* __restrict__ zbf,
                                                    const float* __restrict__ mW3,
                                                    const float* __restrict__ mb3,
                                                    const int* __restrict__ pidx,
                                                    const float* __restrict__ mu_k,
                                                    const float* __restrict__ R_k,
                                                    const float* __restrict__ s_k,
                                                    const float* __restrict__ node_mask,
                                                    float* __restrict__ out) {
    int nn = blockIdx.x * 128 + threadIdx.x;
    if (nn >= Bb * Nn) return;
    float mN = node_mask[nn];
    const unsigned short* z = zbf + (size_t)nn * Hh;
    float a0 = mb3[0], a1 = mb3[1], a2 = mb3[2];
    for (int c = 0; c < Hh; ++c) {
        float zv = bf2f(z[c]);
        a0 += zv * mW3[0 * Hh + c];
        a1 += zv * mW3[1 * Hh + c];
        a2 += zv * mW3[2 * Hh + c];
    }
    float xi0 = tanhf(a0 * mN) * XI_SCALE;
    float xi1 = tanhf(a1 * mN) * XI_SCALE;
    float xi2 = tanhf(a2 * mN) * XI_SCALE;
    int b = nn / Nn;
    int idx = pidx[nn];
    const float* mu = mu_k + (size_t)(b * KP + idx) * 3;
    const float* sk = s_k  + (size_t)(b * KP + idx) * 3;
    const float* R  = R_k  + (size_t)(b * KP + idx) * 9;
    float v0 = xi0 * fmaxf(sk[0] * mN, 1e-6f);
    float v1 = xi1 * fmaxf(sk[1] * mN, 1e-6f);
    float v2 = xi2 * fmaxf(sk[2] * mN, 1e-6f);
    float sc = mN * ANG_SCALE;
    out[(size_t)nn * 3 + 0] = (mu[0] * mN + R[0] * v0 + R[1] * v1 + R[2] * v2) * sc;
    out[(size_t)nn * 3 + 1] = (mu[1] * mN + R[3] * v0 + R[4] * v1 + R[5] * v2) * sc;
    out[(size_t)nn * 3 + 2] = (mu[2] * mN + R[6] * v0 + R[7] * v1 + R[8] * v2) * sc;
}

// ---------------- host ----------------

extern "C" void kernel_launch(void* const* d_in, const int* in_sizes, int n_in,
                              void* d_out, int out_size, void* d_ws, size_t ws_size,
                              hipStream_t stream) {
    const float* s_parent   = (const float*)d_in[0];
    const float* s_trunk    = (const float*)d_in[1];
    const float* A_soft     = (const float*)d_in[2];
    const float* mu_k       = (const float*)d_in[3];
    const float* R_k        = (const float*)d_in[4];
    const float* s_k        = (const float*)d_in[5];
    const float* node_mask  = (const float*)d_in[6];
    const float* mask_parent= (const float*)d_in[7];
    const float* lnq_g = (const float*)d_in[8];
    const float* lnq_b = (const float*)d_in[9];
    const float* lnk_g = (const float*)d_in[10];
    const float* lnk_b = (const float*)d_in[11];
    const float* Wq  = (const float*)d_in[12];
    const float* Wk  = (const float*)d_in[13];
    const float* Wmu = (const float*)d_in[14];
    const float* posW = (const float*)d_in[15];
    const float* posb = (const float*)d_in[16];
    const float* fln_g = (const float*)d_in[17];
    const float* fln_b = (const float*)d_in[18];
    const float* fW1 = (const float*)d_in[19];
    const float* fb1 = (const float*)d_in[20];
    const float* fW2 = (const float*)d_in[21];
    const float* fb2 = (const float*)d_in[22];
    const float* mln_g = (const float*)d_in[23];
    const float* mln_b = (const float*)d_in[24];
    const float* mW1 = (const float*)d_in[25];
    const float* mb1 = (const float*)d_in[26];
    const float* mW2 = (const float*)d_in[27];
    const float* mb2 = (const float*)d_in[28];
    const float* mW3 = (const float*)d_in[29];
    const float* mb3 = (const float*)d_in[30];

    char* ws = (char*)d_ws;
    size_t off = 0;
    auto take = [&](size_t bytes) -> char* {
        char* p = ws + off;
        off += (bytes + 255) & ~(size_t)255;
        return p;
    };
    float*          logits = (float*)         take((size_t)Bb * Nn * KP * 4);   // reused as w_soft
    unsigned short* q_bf   = (unsigned short*)take((size_t)Bb * Nn * Cc * 2);   // also bf16 scratch
    unsigned short* x_bf   = (unsigned short*)take((size_t)Bb * Nn * Cc * 2);
    float*          buf1   = (float*)         take((size_t)Bb * Nn * Cc * 4);   // s_pi
    float*          buf2   = (float*)         take((size_t)Bb * Nn * Cc * 4);   // feat
    float*          k_f32  = (float*)         take((size_t)Bb * KP * Cc * 4);
    unsigned short* k_bf   = (unsigned short*)take((size_t)Bb * KP * Cc * 2);
    unsigned short* w_bf   = (unsigned short*)take((size_t)6 * Cc * Cc * 2);
    int*            pidx   = (int*)           take((size_t)Bb * Nn * 4);
    float*          pos01  = (float*)         take((size_t)Bb * Nn * 4);
    float*          total  = (float*)         take((size_t)Bb * KP * 4);

    unsigned short* Wq_bf  = w_bf + 0 * Cc * Cc;
    unsigned short* Wk_bf  = w_bf + 1 * Cc * Cc;
    unsigned short* fW1_bf = w_bf + 2 * Cc * Cc;
    unsigned short* fW2_bf = w_bf + 3 * Cc * Cc;
    unsigned short* mW1_bf = w_bf + 4 * Cc * Cc;
    unsigned short* mW2_bf = w_bf + 5 * Cc * Cc;

    const int WN = Cc * Cc;
    cast_bf16_kernel<<<WN / 256, 256, 0, stream>>>(Wq,  Wq_bf,  WN);
    cast_bf16_kernel<<<WN / 256, 256, 0, stream>>>(Wk,  Wk_bf,  WN);
    cast_bf16_kernel<<<WN / 256, 256, 0, stream>>>(fW1, fW1_bf, WN);
    cast_bf16_kernel<<<WN / 256, 256, 0, stream>>>(fW2, fW2_bf, WN);
    cast_bf16_kernel<<<WN / 256, 256, 0, stream>>>(mW1, mW1_bf, WN);
    cast_bf16_kernel<<<WN / 256, 256, 0, stream>>>(mW2, mW2_bf, WN);

    // q projection
    ln_cast_kernel<<<Bb * Nn, 256, 0, stream>>>(s_trunk, lnq_g, lnq_b, x_bf);
    wmma_gemm256<0, 1><<<Bb * Nn / 128, 256, 0, stream>>>(x_bf, Wq_bf, nullptr, nullptr, q_bf, Bb * Nn);
    // k projection
    ln_cast_kernel<<<Bb * KP, 256, 0, stream>>>(s_parent, lnk_g, lnk_b, x_bf);
    wmma_gemm256<0, 0><<<Bb * KP / 128, 256, 0, stream>>>(x_bf, Wk_bf, nullptr, k_f32, nullptr, Bb * KP);
    k_finish_kernel<<<Bb * KP, 256, 0, stream>>>(k_f32, mu_k, Wmu, k_bf);
    // attention logits + softmax + argmax
    attn_logits_kernel<<<dim3(Nn / 16 / 8, Bb), 256, 0, stream>>>(q_bf, k_bf, A_soft, node_mask, mask_parent, logits);
    softmax_kernel<<<Bb * Nn / 8, 256, 0, stream>>>(logits, node_mask, pidx);
    // positional scan
    col_total_kernel<<<dim3(KP / 256, Bb), 256, 0, stream>>>(logits, total);
    zero_kernel<<<(Bb * Nn + 255) / 256, 256, 0, stream>>>(pos01, Bb * Nn);
    pos_scan_kernel<<<dim3(KP / 256, Bb), 256, 0, stream>>>(logits, total, pos01);
    // s_pi = gather + pos_emb
    spi_kernel<<<Bb * Nn, 256, 0, stream>>>(pos01, posW, posb, s_parent, pidx, node_mask, buf1);
    // FFN
    ln_cast_kernel<<<Bb * Nn, 256, 0, stream>>>(buf1, fln_g, fln_b, x_bf);
    wmma_gemm256<1, 1><<<Bb * Nn / 128, 256, 0, stream>>>(x_bf, fW1_bf, fb1, nullptr, q_bf, Bb * Nn);
    wmma_gemm256<0, 0><<<Bb * Nn / 128, 256, 0, stream>>>(q_bf, fW2_bf, fb2, buf2, nullptr, Bb * Nn);
    // MLP head
    ln_cast_kernel<<<Bb * Nn, 256, 0, stream>>>(buf2, mln_g, mln_b, x_bf);
    wmma_gemm256<1, 1><<<Bb * Nn / 128, 256, 0, stream>>>(x_bf, mW1_bf, mb1, nullptr, q_bf, Bb * Nn);
    wmma_gemm256<1, 1><<<Bb * Nn / 128, 256, 0, stream>>>(q_bf, mW2_bf, mb2, nullptr, x_bf, Bb * Nn);
    // final gather + rigid transform
    final_kernel<<<(Bb * Nn + 127) / 128, 128, 0, stream>>>(x_bf, mW3, mb3, pidx, mu_k, R_k, s_k, node_mask, (float*)d_out);
}
